// HeteroGNN_103079215236
// MI455X (gfx1250) — matrix operand order repaired and checked by reading, not compiled, not run on hardware
//
#include <hip/hip_runtime.h>

#define D 128
#define TB 256

typedef __attribute__((ext_vector_type(16))) __bf16 bf16x16;
typedef __attribute__((ext_vector_type(8)))  float  v8f;
typedef __attribute__((ext_vector_type(4)))  unsigned int uintx4;

union Frag {
    bf16x16 v;
    uintx4  q[2];
    unsigned int u[8];
};

__device__ __forceinline__ unsigned short f2bf(float f) {
    unsigned u = __float_as_uint(f);
    unsigned r = u + 0x7fffu + ((u >> 16) & 1u);   // round-to-nearest-even
    return (unsigned short)(r >> 16);
}

// ---------------- utility kernels ----------------

__global__ void k_zero(float* __restrict__ p, int n) {
    int t = blockIdx.x * blockDim.x + threadIdx.x;
    if (t < n) p[t] = 0.f;
}

__global__ void k_cvt_bf16(const float* __restrict__ in, unsigned short* __restrict__ out, int n) {
    int t = blockIdx.x * blockDim.x + threadIdx.x;
    if (t < n) out[t] = f2bf(in[t]);
}

// Pack W (optionally W+W2) row-major [128x128] f32 into the per-lane WMMA
// B-fragment order: out[(((ct*4+kidx)*32+lane)*8+v)] = {B[kB][col], B[kB+1][col]}
// with half=lane>>4, l16=lane&15, col=ct*16+l16, kB=kidx*32+half*16+2v.
__global__ void k_pack_b(const float* __restrict__ W, const float* __restrict__ W2,
                         unsigned int* __restrict__ out) {
    int t = blockIdx.x * blockDim.x + threadIdx.x;   // 8192 total
    if (t >= (D * D) / 2) return;
    int v    = t & 7;
    int lane = (t >> 3) & 31;
    int kidx = (t >> 8) & 3;
    int ct   = t >> 10;
    int half = lane >> 4, l16 = lane & 15;
    int col  = ct * 16 + l16;
    int kB   = kidx * 32 + half * 16 + 2 * v;
    float w0 = W[kB * D + col];
    float w1 = W[(kB + 1) * D + col];
    if (W2) { w0 += W2[kB * D + col]; w1 += W2[(kB + 1) * D + col]; }
    out[t] = (unsigned)f2bf(w0) | ((unsigned)f2bf(w1) << 16);
}

// ---------------- WMMA GEMM: C[MxD] (+)= A[MxD](bf16) * Bpacked(bf16) ----------------
// One wave per 16x16 output tile; 8 waves/block cover all 8 column tiles of a
// 16-row stripe. Row index is CLAMPED (not predicated) so EXEC stays all-ones
// through the whole K loop: body is pure clause'd b128 loads + 4 WMMAs.
// ACC is a compile-time flag; the store epilogue has a uniform full-tile fast
// path (no per-row guards) and a guarded generic tail path.

template <bool ACC>
__global__ void k_gemm_bf16(const unsigned short* __restrict__ A,
                            const unsigned int* __restrict__ Bp,
                            float* __restrict__ C, int M) {
    int wave = threadIdx.x >> 5;            // 0..7 -> column tile
    int lane = threadIdx.x & 31;
    int half = lane >> 4, l16 = lane & 15;
    int row0 = blockIdx.x * 16;
    int col0 = wave * 16;
    int arow = min(row0 + l16, M - 1);      // branchless clamp (dup last row for tail)
    const unsigned int* A32 = (const unsigned int*)A;   // bf16 pairs

    v8f acc = {};
    #pragma unroll
    for (int kidx = 0; kidx < 4; ++kidx) {
        Frag a, b;
        int abase = arow * (D / 2) + kidx * 16 + half * 8;
        a.q[0] = *(const uintx4*)(A32 + abase);
        a.q[1] = *(const uintx4*)(A32 + abase + 4);
        const unsigned int* bp = Bp + (((wave * 4 + kidx) * 32 + lane) * 8);
        b.q[0] = *(const uintx4*)(bp);
        b.q[1] = *(const uintx4*)(bp + 4);
        acc = __builtin_amdgcn_wmma_f32_16x16x32_bf16(
            false, a.v, false, b.v, (short)0, acc, false, false);
    }

    float* cp = C + (row0 + half * 8) * D + col0 + l16;   // row stride D floats
    if (row0 + 16 <= M) {                 // uniform fast path: full tile
        #pragma unroll
        for (int r = 0; r < 8; ++r) {
            if (ACC) cp[r * D] += acc[r]; else cp[r * D] = acc[r];
        }
    } else {                              // generic tail (unused when M % 16 == 0)
        #pragma unroll
        for (int r = 0; r < 8; ++r) {
            if (row0 + half * 8 + r < M) {
                if (ACC) cp[r * D] += acc[r]; else cp[r * D] = acc[r];
            }
        }
    }
}

// ---------------- graph kernels ----------------

__global__ void k_degree(const int* __restrict__ dst, float* __restrict__ cnt, int E) {
    int t = blockIdx.x * blockDim.x + threadIdx.x;
    if (t < E) atomicAdd(&cnt[dst[t]], 1.f);
}

__global__ void k_recip(float* __restrict__ cnt, int n) {
    int t = blockIdx.x * blockDim.x + threadIdx.x;
    if (t < n) cnt[t] = 1.f / fmaxf(cnt[t], 1.f);
}

// one wave per edge: agg[dst] += h[src] * rcnt[dst]
__global__ void k_scatter(const float* __restrict__ h, const int* __restrict__ src,
                          const int* __restrict__ dst, const float* __restrict__ rcnt,
                          float* __restrict__ agg, int E) {
    int t = blockIdx.x * blockDim.x + threadIdx.x;
    int e = t >> 5, lane = t & 31;
    if (e >= E) return;
    int s = src[e], d = dst[e];
    float r = rcnt[d];
    float4 v = *(const float4*)(h + s * D + lane * 4);
    float* p = agg + d * D + lane * 4;
    atomicAdd(p + 0, v.x * r);
    atomicAdd(p + 1, v.y * r);
    atomicAdd(p + 2, v.z * r);
    atomicAdd(p + 3, v.w * r);
}

// per-column sum / sumsq -> sums[0..127]=sum, sums[128..255]=sumsq
__global__ void k_stats(const float* __restrict__ h, float* __restrict__ sums, int Nn) {
    __shared__ float ls[2 * D];
    ls[threadIdx.x] = 0.f;
    __syncthreads();
    int t = blockIdx.x * blockDim.x + threadIdx.x;
    int node = t >> 5, lane = t & 31;
    if (node < Nn) {
        float4 v = *(const float4*)(h + node * D + lane * 4);
        int c = lane * 4;
        atomicAdd(&ls[c + 0], v.x); atomicAdd(&ls[c + 1], v.y);
        atomicAdd(&ls[c + 2], v.z); atomicAdd(&ls[c + 3], v.w);
        atomicAdd(&ls[D + c + 0], v.x * v.x); atomicAdd(&ls[D + c + 1], v.y * v.y);
        atomicAdd(&ls[D + c + 2], v.z * v.z); atomicAdd(&ls[D + c + 3], v.w * v.w);
    }
    __syncthreads();
    atomicAdd(&sums[threadIdx.x], ls[threadIdx.x]);
}

__global__ void k_bn_finalize(const float* __restrict__ sums, const float* __restrict__ g,
                              const float* __restrict__ b, float* __restrict__ sb, float Ninv) {
    int c = threadIdx.x;              // 128 threads
    float mu  = sums[c] * Ninv;
    float var = fmaxf(sums[D + c] * Ninv - mu * mu, 0.f);
    float sc  = g[c] * rsqrtf(var + 1e-5f);
    sb[c]     = sc;
    sb[D + c] = b[c] - mu * sc;
}

__global__ void k_bn_apply(const float* __restrict__ h, const float* __restrict__ sb,
                           float* __restrict__ outf, unsigned short* __restrict__ outb,
                           int total, int leaky) {
    int t = blockIdx.x * blockDim.x + threadIdx.x;
    if (t >= total) return;
    int c = t & (D - 1);
    float y = h[t] * sb[c] + sb[D + c];
    if (leaky && y < 0.f) y *= 0.01f;
    if (outf) outf[t] = y;
    if (outb) outb[t] = f2bf(y);
}

// one wave per label edge: out[e] = dot(h[a[e]], h[b[e]])
__global__ void k_edge_dot(const float* __restrict__ h, const int* __restrict__ a,
                           const int* __restrict__ b, float* __restrict__ out, int EL) {
    int t = blockIdx.x * blockDim.x + threadIdx.x;
    int e = t >> 5, lane = t & 31;
    if (e >= EL) return;
    float4 va = *(const float4*)(h + a[e] * D + lane * 4);
    float4 vb = *(const float4*)(h + b[e] * D + lane * 4);
    float s = va.x * vb.x + va.y * vb.y + va.z * vb.z + va.w * vb.w;
    #pragma unroll
    for (int o = 16; o > 0; o >>= 1) s += __shfl_xor(s, o, 32);
    if (lane == 0) out[e] = s;
}

// ---------------- host orchestration ----------------

static inline int cdiv(long long a, long long b) { return (int)((a + b - 1) / b); }

extern "C" void kernel_launch(void* const* d_in, const int* in_sizes, int n_in,
                              void* d_out, int out_size, void* d_ws, size_t ws_size,
                              hipStream_t stream) {
    const float* x    = (const float*)d_in[0];
    const int*   ei1  = (const int*)d_in[1];
    const int*   ei2  = (const int*)d_in[2];
    const int*   eli  = (const int*)d_in[3];
    const float* W1n1 = (const float*)d_in[4];
    const float* W1s1 = (const float*)d_in[5];
    const float* W1n2 = (const float*)d_in[6];
    const float* W1s2 = (const float*)d_in[7];
    const float* W2n1 = (const float*)d_in[8];
    const float* W2s1 = (const float*)d_in[9];
    const float* W2n2 = (const float*)d_in[10];
    const float* W2s2 = (const float*)d_in[11];
    const float* g1   = (const float*)d_in[12];
    const float* b1   = (const float*)d_in[13];
    const float* g2   = (const float*)d_in[14];
    const float* b2   = (const float*)d_in[15];

    const int N  = in_sizes[0] / D;
    const int E1 = in_sizes[1] / 2;
    const int E2 = in_sizes[2] / 2;
    const int EL = in_sizes[3] / 2;
    const int ND = N * D;

    // workspace carve-up (256B aligned)
    char* w = (char*)d_ws;
    auto carve = [&](size_t bytes) { void* p = (void*)w; w += (bytes + 255) & ~(size_t)255; return p; };
    float*          bufNbr = (float*)carve((size_t)ND * 4);
    float*          agg    = (float*)carve((size_t)ND * 4);
    unsigned short* xb     = (unsigned short*)carve((size_t)ND * 2);
    unsigned int*   wpack  = (unsigned int*)carve((size_t)6 * (D * D / 2) * 4);
    float*          rc1    = (float*)carve((size_t)N * 4);
    float*          rc2    = (float*)carve((size_t)N * 4);
    float*          stats  = (float*)carve(2 * D * 4);
    float*          sb     = (float*)carve(2 * D * 4);

    const int*  src1 = ei1,  *dst1 = ei1 + E1;
    const int*  src2 = ei2,  *dst2 = ei2 + E2;
    const int*  elA  = eli,  *elB  = eli + EL;
    unsigned int* wp[6] = { wpack, wpack + 8192, wpack + 2 * 8192,
                            wpack + 3 * 8192, wpack + 4 * 8192, wpack + 5 * 8192 };

    // --- one-time prep (per call) ---
    k_zero<<<cdiv(N, TB), TB, 0, stream>>>(rc1, N);
    k_zero<<<cdiv(N, TB), TB, 0, stream>>>(rc2, N);
    k_degree<<<cdiv(E1, TB), TB, 0, stream>>>(dst1, rc1, E1);
    k_degree<<<cdiv(E2, TB), TB, 0, stream>>>(dst2, rc2, E2);
    k_recip<<<cdiv(N, TB), TB, 0, stream>>>(rc1, N);
    k_recip<<<cdiv(N, TB), TB, 0, stream>>>(rc2, N);
    k_cvt_bf16<<<cdiv(ND, TB), TB, 0, stream>>>(x, xb, ND);
    k_pack_b<<<32, TB, 0, stream>>>(W1n1, nullptr, wp[0]);
    k_pack_b<<<32, TB, 0, stream>>>(W1n2, nullptr, wp[1]);
    k_pack_b<<<32, TB, 0, stream>>>(W1s1, W1s2,    wp[2]);
    k_pack_b<<<32, TB, 0, stream>>>(W2n1, nullptr, wp[3]);
    k_pack_b<<<32, TB, 0, stream>>>(W2n2, nullptr, wp[4]);
    k_pack_b<<<32, TB, 0, stream>>>(W2s1, W2s2,    wp[5]);

    const int gemmGrid = cdiv(N, 16);
    const float Ninv = 1.f / (float)N;

    // ---------------- layer 1 ----------------
    k_zero<<<cdiv(ND, TB), TB, 0, stream>>>(agg, ND);
    k_zero<<<1, TB, 0, stream>>>(stats, 2 * D);
    k_gemm_bf16<false><<<gemmGrid, TB, 0, stream>>>(xb, wp[0], bufNbr, N);
    k_scatter<<<cdiv((long long)E1 * 32, TB), TB, 0, stream>>>(bufNbr, src1, dst1, rc1, agg, E1);
    k_gemm_bf16<false><<<gemmGrid, TB, 0, stream>>>(xb, wp[1], bufNbr, N);
    k_scatter<<<cdiv((long long)E2 * 32, TB), TB, 0, stream>>>(bufNbr, src2, dst2, rc2, agg, E2);
    k_gemm_bf16<true><<<gemmGrid, TB, 0, stream>>>(xb, wp[2], agg, N);   // += x @ (Ws1+Ws2)
    k_stats<<<cdiv((long long)N * 32, TB), TB, 0, stream>>>(agg, stats, N);
    k_bn_finalize<<<1, D, 0, stream>>>(stats, g1, b1, sb, Ninv);
    k_bn_apply<<<cdiv(ND, TB), TB, 0, stream>>>(agg, sb, nullptr, xb, ND, 1); // leaky, bf16 h1

    // ---------------- layer 2 ----------------
    k_zero<<<cdiv(ND, TB), TB, 0, stream>>>(agg, ND);
    k_zero<<<1, TB, 0, stream>>>(stats, 2 * D);
    k_gemm_bf16<false><<<gemmGrid, TB, 0, stream>>>(xb, wp[3], bufNbr, N);
    k_scatter<<<cdiv((long long)E1 * 32, TB), TB, 0, stream>>>(bufNbr, src1, dst1, rc1, agg, E1);
    k_gemm_bf16<false><<<gemmGrid, TB, 0, stream>>>(xb, wp[4], bufNbr, N);
    k_scatter<<<cdiv((long long)E2 * 32, TB), TB, 0, stream>>>(bufNbr, src2, dst2, rc2, agg, E2);
    k_gemm_bf16<true><<<gemmGrid, TB, 0, stream>>>(xb, wp[5], agg, N);   // += h1 @ (Ws1+Ws2)
    k_stats<<<cdiv((long long)N * 32, TB), TB, 0, stream>>>(agg, stats, N);
    k_bn_finalize<<<1, D, 0, stream>>>(stats, g2, b2, sb, Ninv);
    k_bn_apply<<<cdiv(ND, TB), TB, 0, stream>>>(agg, sb, bufNbr, nullptr, ND, 0); // h2 f32

    // ---------------- edge-label dot product ----------------
    k_edge_dot<<<cdiv((long long)EL * 32, TB), TB, 0, stream>>>(bufNbr, elA, elB, (float*)d_out, EL);
}
